// BidirectionalGINConv_19610820673951
// MI455X (gfx1250) — matrix operand order repaired and checked by reading, not compiled, not run on hardware
//
#include <hip/hip_runtime.h>

#define N_NODES 50000
#define N_EDGES 800000
#define D 128

typedef __bf16 bf16;
typedef __attribute__((ext_vector_type(16))) __bf16 v16bf;
typedef __attribute__((ext_vector_type(8)))  float  v8f;

__device__ __forceinline__ void split_bf16(float x, bf16& hi, bf16& lo) {
    hi = (bf16)x;
    lo = (bf16)(x - (float)hi);
}

// ---------------------------------------------------------------------------
// Kernel 1: agg_a = agg_b = x  (so h0 = x + neighbor_sum falls out of scatter)
//           and split w1/w2 into bf16 hi/lo copies (128KB, L2-resident).
// ---------------------------------------------------------------------------
__global__ void gin_prep_kernel(const float* __restrict__ x,
                                const float* __restrict__ w1,
                                const float* __restrict__ w2,
                                float* __restrict__ agg_a,
                                float* __restrict__ agg_b,
                                bf16* __restrict__ w1h, bf16* __restrict__ w1l,
                                bf16* __restrict__ w2h, bf16* __restrict__ w2l) {
    long i = (long)blockIdx.x * blockDim.x + threadIdx.x;
    if (i < (long)N_NODES * D) {
        float v = x[i];
        agg_a[i] = v;
        agg_b[i] = v;
    }
    if (i < (long)D * D) {
        bf16 h, l;
        float a = w1[i]; split_bf16(a, h, l); w1h[i] = h; w1l[i] = l;
        float b = w2[i]; split_bf16(b, h, l); w2h[i] = h; w2l[i] = l;
    }
}

// ---------------------------------------------------------------------------
// Kernel 2: edge scatter. One wave per (edge, direction); each lane gathers a
// float4 of x[src] and does 4 hardware f32 atomics into agg[dst]. Both x and
// agg live in L2 (25.6MB each), so this is L2-atomic bound, not HBM bound.
// ---------------------------------------------------------------------------
__global__ void gin_scatter_kernel(const float* __restrict__ x,
                                   const int* __restrict__ ei,
                                   const int* __restrict__ rev,
                                   float* __restrict__ agg_a,
                                   float* __restrict__ agg_b) {
    unsigned tid  = blockIdx.x * blockDim.x + threadIdx.x;
    unsigned wid  = tid >> 5;   // global wave id (wave32)
    unsigned lane = tid & 31;
    if (wid >= 2u * N_EDGES) return;

    const int* idx;
    float* agg;
    unsigned e;
    if (wid < N_EDGES) { idx = ei;  agg = agg_a; e = wid; }
    else               { idx = rev; agg = agg_b; e = wid - N_EDGES; }

    int src = idx[e];             // edge_index[0, e]
    int dst = idx[N_EDGES + e];   // edge_index[1, e]

    const float4 v = *(const float4*)(x + (long)src * D + lane * 4);
    float* p = agg + (long)dst * D + lane * 4;
    unsafeAtomicAdd(p + 0, v.x);
    unsafeAtomicAdd(p + 1, v.y);
    unsafeAtomicAdd(p + 2, v.z);
    unsafeAtomicAdd(p + 3, v.w);
}

// ---------------------------------------------------------------------------
// Kernel 3: fused 2-layer MLP for both directions + combine + final relu.
// Block = 256 threads = 8 waves; block owns 16 nodes; wave w owns the 16x16
// output tile covering columns [16w, 16w+16). bf16x3 split GEMM via
// v_wmma_f32_16x16x32_bf16 (f32 accumulate), h1 stays in LDS.
// ---------------------------------------------------------------------------
__global__ __launch_bounds__(256) void gin_mlp_kernel(
    const float* __restrict__ agg_a, const float* __restrict__ agg_b,
    const bf16* __restrict__ w1h, const bf16* __restrict__ w1l,
    const bf16* __restrict__ w2h, const bf16* __restrict__ w2l,
    const float* __restrict__ b1, const float* __restrict__ b2,
    float* __restrict__ out) {

    __shared__ bf16 sAh[2][16][D];   // h0 hi, both directions
    __shared__ bf16 sAl[2][16][D];   // h0 lo
    __shared__ bf16 sHh[2][16][D];   // h1 hi
    __shared__ bf16 sHl[2][16][D];   // h1 lo

    const int  tid  = threadIdx.x;
    const int  wave = tid >> 5;     // 0..7  -> column tile
    const int  lane = tid & 31;
    const int  c    = lane & 15;    // row (A) / col (B,D) within tile
    const int  half = lane >> 4;
    const int  j0   = wave * 16;    // output-column base of this wave's tile
    const long node0 = (long)blockIdx.x * 16;

    // Stage h0 = x + agg for both directions into LDS as bf16 hi/lo.
    for (int i = tid; i < 16 * D; i += 256) {
        int m = i >> 7, col = i & (D - 1);
        bf16 h, l;
        float va = agg_a[(node0 + m) * D + col];
        split_bf16(va, h, l); sAh[0][m][col] = h; sAl[0][m][col] = l;
        float vb = agg_b[(node0 + m) * D + col];
        split_bf16(vb, h, l); sAh[1][m][col] = h; sAl[1][m][col] = l;
    }
    __syncthreads();

    const float bias1 = b1[j0 + c];

    // ---- Layer 1: h1 = relu(h0 @ w1^T + b1), both directions -> LDS ----
    for (int d = 0; d < 2; ++d) {
        v8f acc = {};
        for (int kb = 0; kb < D; kb += 32) {
            v16bf ah, al, bh, bl;
            // A frag (16x32 bf16): element e -> K = half*16 + e
            #pragma unroll
            for (int e = 0; e < 16; ++e) {
                int k = kb + half * 16 + e;
                ah[e] = sAh[d][c][k];
                al[e] = sAl[d][c][k];
            }
            // B frag (32x16 bf16): B[k, j] = w1[j, k]; element e -> K = half*16 + e
            const bf16* wrh = w1h + (j0 + c) * D + kb + half * 16;
            const bf16* wrl = w1l + (j0 + c) * D + kb + half * 16;
            #pragma unroll
            for (int e = 0; e < 16; ++e) { bh[e] = wrh[e]; bl[e] = wrl[e]; }
            acc = __builtin_amdgcn_wmma_f32_16x16x32_bf16(false, ah, false, bh, (short)0, acc, false, false);
            acc = __builtin_amdgcn_wmma_f32_16x16x32_bf16(false, ah, false, bl, (short)0, acc, false, false);
            acc = __builtin_amdgcn_wmma_f32_16x16x32_bf16(false, al, false, bh, (short)0, acc, false, false);
        }
        #pragma unroll
        for (int r = 0; r < 8; ++r) {   // D tile: vgpr r -> row r + half*8, col c
            float v = acc[r] + bias1;
            v = v > 0.f ? v : 0.f;
            bf16 h, l; split_bf16(v, h, l);
            int m = r + half * 8;
            sHh[d][m][j0 + c] = h;
            sHl[d][m][j0 + c] = l;
        }
    }
    __syncthreads();

    // ---- Layer 2: o = h1 @ w2^T; out = relu(0.5*(o_a + o_b) + b2) ----
    const float bias2 = b2[j0 + c];
    v8f acc2[2];
    for (int d = 0; d < 2; ++d) {
        v8f acc = {};
        for (int kb = 0; kb < D; kb += 32) {
            v16bf ah, al, bh, bl;
            #pragma unroll
            for (int e = 0; e < 16; ++e) {
                int k = kb + half * 16 + e;
                ah[e] = sHh[d][c][k];
                al[e] = sHl[d][c][k];
            }
            const bf16* wrh = w2h + (j0 + c) * D + kb + half * 16;
            const bf16* wrl = w2l + (j0 + c) * D + kb + half * 16;
            #pragma unroll
            for (int e = 0; e < 16; ++e) { bh[e] = wrh[e]; bl[e] = wrl[e]; }
            acc = __builtin_amdgcn_wmma_f32_16x16x32_bf16(false, ah, false, bh, (short)0, acc, false, false);
            acc = __builtin_amdgcn_wmma_f32_16x16x32_bf16(false, ah, false, bl, (short)0, acc, false, false);
            acc = __builtin_amdgcn_wmma_f32_16x16x32_bf16(false, al, false, bh, (short)0, acc, false, false);
        }
        acc2[d] = acc;
    }
    #pragma unroll
    for (int r = 0; r < 8; ++r) {
        float v = 0.5f * (acc2[0][r] + acc2[1][r]) + bias2;
        v = v > 0.f ? v : 0.f;
        int m = r + half * 8;
        out[(node0 + m) * D + j0 + c] = v;
    }
}

// ---------------------------------------------------------------------------
extern "C" void kernel_launch(void* const* d_in, const int* in_sizes, int n_in,
                              void* d_out, int out_size, void* d_ws, size_t ws_size,
                              hipStream_t stream) {
    const float* x   = (const float*)d_in[0];
    const int*   ei  = (const int*)d_in[1];
    const int*   rev = (const int*)d_in[2];
    const float* w1  = (const float*)d_in[3];
    const float* b1  = (const float*)d_in[4];
    const float* w2  = (const float*)d_in[5];
    const float* b2  = (const float*)d_in[6];
    float* out = (float*)d_out;

    // Workspace carve-up (~51.3 MB total)
    char* ws = (char*)d_ws;
    size_t off = 0;
    auto alloc = [&](size_t bytes) {
        void* p = ws + off;
        off = (off + bytes + 255) & ~(size_t)255;
        return p;
    };
    float* agg_a = (float*)alloc(sizeof(float) * (size_t)N_NODES * D);
    float* agg_b = (float*)alloc(sizeof(float) * (size_t)N_NODES * D);
    bf16*  w1h   = (bf16*)alloc(sizeof(bf16) * D * D);
    bf16*  w1l   = (bf16*)alloc(sizeof(bf16) * D * D);
    bf16*  w2h   = (bf16*)alloc(sizeof(bf16) * D * D);
    bf16*  w2l   = (bf16*)alloc(sizeof(bf16) * D * D);

    // 1) init agg buffers with x, split weights to bf16 hi/lo
    {
        long total = (long)N_NODES * D;                 // 6.4M
        int blocks = (int)((total + 255) / 256);
        gin_prep_kernel<<<blocks, 256, 0, stream>>>(x, w1, w2, agg_a, agg_b,
                                                    w1h, w1l, w2h, w2l);
    }
    // 2) edge scatter: one wave per (edge, direction)
    {
        long waves   = 2L * N_EDGES;                    // 1.6M waves
        long threads = waves * 32;
        int blocks = (int)((threads + 255) / 256);      // 200000
        gin_scatter_kernel<<<blocks, 256, 0, stream>>>(x, ei, rev, agg_a, agg_b);
    }
    // 3) fused MLP + combine
    {
        int blocks = N_NODES / 16;                      // 3125 (exact)
        gin_mlp_kernel<<<blocks, 256, 0, stream>>>(agg_a, agg_b,
                                                   w1h, w1l, w2h, w2l,
                                                   b1, b2, out);
    }
}
